// NSDEModel_10015863734397
// MI455X (gfx1250) — matrix operand order repaired and checked by reading, not compiled, not run on hardware
//
#include <hip/hip_runtime.h>
#include <math.h>

// Problem constants (from reference)
#define B_TOTAL 4096
#define T_STEPS 256
#define F_IN    32
#define H_DIM   64
#define DT_C    0.1f
#define SQRT_DT 0.31622776601683794f
#define NW      4                 // waves per workgroup (wave32)
#define BLOCK_THREADS (NW * 32)
#define ROWS_PER_WG (NW * 16)     // 64 rows per workgroup

typedef __attribute__((ext_vector_type(16))) __bf16        v16bf;
typedef __attribute__((ext_vector_type(8)))  float         v8f;
typedef __attribute__((ext_vector_type(4)))  unsigned int  v4u;
typedef __attribute__((ext_vector_type(4)))  unsigned int  u32x4;
typedef __attribute__((ext_vector_type(8)))  unsigned int  u32x8;

union ABits { v16bf v; v4u q[2]; __bf16 h[16]; };

// LDS layout (bytes). All weight blocks stored bf16 column-major: w[n*K + k].
enum : int {
  OFF_W1   = 0,       // 96x128  bf16 cm : 24576
  OFF_W2   = 24576,   // 128x128 bf16 cm : 32768
  OFF_W3   = 57344,   // 128x64  bf16 cm : 16384
  OFF_D1   = 73728,   // 64x64   bf16 cm : 8192
  OFF_D2   = 81920,   // 64x64   bf16 cm : 8192
  OFF_R1   = 90112,   // 64x32   bf16 cm : 4096
  OFF_R2   = 94208,   // 32x16   bf16 cm (N padded 2->16 with zeros) : 1024
  OFF_B1   = 95232,   // 128 f32
  OFF_B2   = 95744,   // 128 f32
  OFF_B3   = 96256,   // 64 f32
  OFF_D1B  = 96512,   // 64 f32
  OFF_D2B  = 96768,   // 64 f32
  OFF_SC   = 97024,   // 64 f32
  OFF_R1B  = 97280,   // 32 f32
  OFF_R2B  = 97408,   // 16 f32 (padded)
  OFF_STAGE= 97472,   // NW * (16*128 bf16 = 4096 B) per-wave act staging
  OFF_NZ   = OFF_STAGE + NW * 4096,   // NW * (16*64 f32 = 4096 B) TDM noise tiles
  SMEM_BYTES = OFF_NZ + NW * 4096     // 130240 bytes
};

__device__ __forceinline__ v8f zero8() {
  v8f z = {0.f,0.f,0.f,0.f,0.f,0.f,0.f,0.f};
  return z;
}

__device__ __forceinline__ v8f wmma_bf16(v16bf a, v16bf b, v8f c) {
  // D = A(16x32 bf16) x B(32x16 bf16) + C(16x16 f32)
  return __builtin_amdgcn_wmma_f32_16x16x32_bf16(false, a, false, b, (short)0, c,
                                                 false, false);
}

// B-operand tile from column-major LDS weights.
// ISA B layout (32x16 bf16): lane holds column n = lane&15; element e holds
// K = kc*32 + 8*(lane>>4) + e (e<8), K = kc*32 + 16 + 8*(lane>>4) + (e-8) (e>=8).
__device__ __forceinline__ v16bf ldsB(const __bf16* w, int K, int nt, int kc, int lane) {
  int n = lane & 15, hk = lane >> 4;
  const __bf16* p = w + (nt * 16 + n) * K + kc * 32 + hk * 8;
  ABits a;
  a.q[0] = *(const v4u*)(p);
  a.q[1] = *(const v4u*)(p + 16);
  return a.v;
}

// A-operand tile from row-major bf16 staging (row stride Wd elements).
// ISA A layout (16x32 bf16): lane row m = lane&15; element e:
//   e in [0,7]  -> K = kc*32 + 8*hk + e
//   e in [8,15] -> K = kc*32 + 16 + 8*hk + (e-8)
__device__ __forceinline__ v16bf ldsA(const __bf16* st, int Wd, int kc, int lane) {
  int m = lane & 15, hk = lane >> 4;
  const __bf16* p = st + m * Wd + kc * 32 + hk * 8;
  ABits a;
  a.q[0] = *(const v4u*)(p);        // K = kc*32 + 8hk .. +7
  a.q[1] = *(const v4u*)(p + 16);   // K = kc*32 + 16 + 8hk .. +7
  return a.v;
}

// Store a 16x16 f32 C tile (optionally +bias, relu) to bf16 row-major staging.
// ISA C/D layout: lane L holds N = L&15, rows M = v + 8*(L>>4) for VGPR v.
__device__ __forceinline__ void stC(__bf16* st, int Wd, int nt, v8f c, int lane,
                                    bool do_relu, const float* bias) {
  int n = lane & 15, mh = lane >> 4;
  float b = bias ? bias[nt * 16 + n] : 0.0f;
#pragma unroll
  for (int v = 0; v < 8; ++v) {
    float x = c[v] + b;
    if (do_relu) x = fmaxf(x, 0.0f);
    st[(v + 8 * mh) * Wd + nt * 16 + n] = (__bf16)x;    // native v_cvt bf16
  }
}

__device__ __forceinline__ void cvt_w(const float* g, __bf16* l, int K, int N,
                                      int tid, int nth) {
  for (int i = tid; i < K * N; i += nth) {
    int k = i / N, n = i % N;                 // g is row-major (K,N)
    l[n * K + k] = (__bf16)g[i];              // store column-major
  }
}

__device__ __forceinline__ void cvt_b(const float* g, float* l, int N, int tid, int nth) {
  for (int i = tid; i < N; i += nth) l[i] = g[i];
}

__global__ __launch_bounds__(BLOCK_THREADS)
void nsde_fused_kernel(const float* __restrict__ x_path,   // (B,T,F)
                       const float* __restrict__ noise,    // (T-1,B,H)
                       const float* __restrict__ W1, const float* __restrict__ b1,
                       const float* __restrict__ W2, const float* __restrict__ b2,
                       const float* __restrict__ W3, const float* __restrict__ b3,
                       const float* __restrict__ D1, const float* __restrict__ d1,
                       const float* __restrict__ D2, const float* __restrict__ d2,
                       const float* __restrict__ scale,
                       const float* __restrict__ R1, const float* __restrict__ r1,
                       const float* __restrict__ R2, const float* __restrict__ r2,
                       float* __restrict__ out)             // (2*B) flat: col0 then col1
{
  extern __shared__ char dyn_smem[];
  __bf16* lds_w1  = (__bf16*)(dyn_smem + OFF_W1);
  __bf16* lds_w2  = (__bf16*)(dyn_smem + OFF_W2);
  __bf16* lds_w3  = (__bf16*)(dyn_smem + OFF_W3);
  __bf16* lds_d1w = (__bf16*)(dyn_smem + OFF_D1);
  __bf16* lds_d2w = (__bf16*)(dyn_smem + OFF_D2);
  __bf16* lds_r1w = (__bf16*)(dyn_smem + OFF_R1);
  __bf16* lds_r2w = (__bf16*)(dyn_smem + OFF_R2);
  float* lds_b1  = (float*)(dyn_smem + OFF_B1);
  float* lds_b2  = (float*)(dyn_smem + OFF_B2);
  float* lds_b3  = (float*)(dyn_smem + OFF_B3);
  float* lds_d1b = (float*)(dyn_smem + OFF_D1B);
  float* lds_d2b = (float*)(dyn_smem + OFF_D2B);
  float* lds_sc  = (float*)(dyn_smem + OFF_SC);
  float* lds_r1b = (float*)(dyn_smem + OFF_R1B);
  float* lds_r2b = (float*)(dyn_smem + OFF_R2B);

  const int tid = threadIdx.x, nth = blockDim.x;

  // ---- one-time: weights f32 -> bf16 column-major in LDS ----
  cvt_w(W1, lds_w1, 96, 128, tid, nth);
  cvt_w(W2, lds_w2, 128, 128, tid, nth);
  cvt_w(W3, lds_w3, 128, 64, tid, nth);
  cvt_w(D1, lds_d1w, 64, 64, tid, nth);
  cvt_w(D2, lds_d2w, 64, 64, tid, nth);
  cvt_w(R1, lds_r1w, 64, 32, tid, nth);
  for (int i = tid; i < 32 * 16; i += nth) lds_r2w[i] = (__bf16)0.0f;  // pad N=2 -> 16
  __syncthreads();
  for (int i = tid; i < 32 * 2; i += nth) {
    int k = i >> 1, n = i & 1;
    lds_r2w[n * 32 + k] = (__bf16)R2[i];
  }
  cvt_b(b1, lds_b1, 128, tid, nth);
  cvt_b(b2, lds_b2, 128, tid, nth);
  cvt_b(b3, lds_b3, 64, tid, nth);
  cvt_b(d1, lds_d1b, 64, tid, nth);
  cvt_b(d2, lds_d2b, 64, tid, nth);
  cvt_b(scale, lds_sc, 64, tid, nth);
  cvt_b(r1, lds_r1b, 32, tid, nth);
  for (int i = tid; i < 16; i += nth) lds_r2b[i] = (i < 2) ? r2[i] : 0.0f;
  __syncthreads();

  // ---- per-wave tile ----
  const int lane = tid & 31;
  const int w    = tid >> 5;
  const int n    = lane & 15;     // C-layout column within N-tile
  const int hk   = lane >> 4;     // C-layout row-half / A,B K-half
  const int r0   = blockIdx.x * ROWS_PER_WG + w * 16;   // first batch row of tile
  __bf16* stage  = (__bf16*)(dyn_smem + OFF_STAGE) + w * 2048;  // 16x128 bf16
  const float* nzlds = (const float*)(dyn_smem + OFF_NZ) + w * 1024; // 16x64 f32
  const float* xb = x_path + (size_t)r0 * T_STEPS * F_IN;

  // Workgroup-relative LDS byte offset of this wave's noise buffer
  // (flat->LDS aperture mapping: low 32 bits of the generic address).
  const unsigned nz_lds_off = (unsigned)(uintptr_t)(const void*)nzlds;
  // Uniform global base of this wave's noise rows (16 rows = 4 KB contiguous).
  const unsigned long long nz_ga0 =
      (unsigned long long)(uintptr_t)noise + (unsigned long long)r0 * H_DIM * 4ull;

  // TDM D# group 1 (constant): mask=0, data_size=4B, 1-D tile of 1024 elems.
  u32x8 g1;
  g1[0] = 0x00020000u;          // workgroup_mask=0 | data_size=2 (4B) << 16
  g1[1] = 1024u << 16;          // tensor_dim0[15:0] = 1024
  g1[2] = 1u << 16;             // tensor_dim0[31:16]=0 | tensor_dim1[15:0]=1
  g1[3] = 1024u << 16;          // tensor_dim1[31:16]=0 | tile_dim0 = 1024
  g1[4] = 0u;                   // tile_dim1 = 0 (unused), tile_dim2 = 0
  g1[5] = 1024u;                // tensor_dim0_stride[31:0] = 1024
  g1[6] = 0u;
  g1[7] = 0u;

  v8f hc[4];                    // h state, f32 C-layout (16 rows x 64 cols)
#pragma unroll
  for (int nt = 0; nt < 4; ++nt) hc[nt] = zero8();

  // ---- Euler–Maruyama scan: 255 steps, fully fused ----
  for (int step = 0; step < T_STEPS - 1; ++step) {
    // Kick off the TDM copy of this step's noise tile (4 KB) into LDS; it
    // runs behind the three drift GEMMs. Guard the buffer against last
    // step's still-pending ds reads first.
    {
      unsigned long long ga = nz_ga0 + (unsigned long long)step * (B_TOTAL * H_DIM * 4ull);
      u32x4 g0;
      g0[0] = 1u;                                   // count=1, user descriptor
      g0[1] = nz_lds_off;                           // lds_addr
      g0[2] = (unsigned)ga;                         // global_addr[31:0]
      g0[3] = (unsigned)((ga >> 32) & 0x01FFFFFFu)  // global_addr[56:32]
              | 0x80000000u;                        // type=2 ("image")
      asm volatile("s_wait_dscnt 0x0" ::: "memory");
      asm volatile("tensor_load_to_lds %0, %1" :: "s"(g0), "s"(g1) : "memory");
    }
    if (step + 1 < T_STEPS - 1) {
      __builtin_prefetch(xb + (size_t)(lane & 15) * (T_STEPS * F_IN) + (step + 1) * F_IN, 0, 3);
    }

    // h (f32 C-layout) -> bf16 staging -> A tiles for K=0..63
#pragma unroll
    for (int nt = 0; nt < 4; ++nt) stC(stage, 64, nt, hc[nt], lane, false, nullptr);
    asm volatile("" ::: "memory");
    v16bf hA0 = ldsA(stage, 64, 0, lane);
    v16bf hA1 = ldsA(stage, 64, 1, lane);

    // x_t A tile (16x32), direct from global, native cvt f32->bf16 in regs
    ABits xa;
    const float* xp = xb + (size_t)(lane & 15) * (T_STEPS * F_IN) + step * F_IN + hk * 8;
#pragma unroll
    for (int j = 0; j < 8; ++j) {
      xa.h[j]     = (__bf16)xp[j];        // K = 8hk + j
      xa.h[8 + j] = (__bf16)xp[16 + j];   // K = 16 + 8hk + j
    }
    v16bf xA = xa.v;

    // GEMM1: f1 = z @ W1   (M16, N128, K96 -> 24 wmma)
    v8f f1[8];
#pragma unroll
    for (int nt = 0; nt < 8; ++nt) {
      v8f c = zero8();
      c = wmma_bf16(hA0, ldsB(lds_w1, 96, nt, 0, lane), c);   // h cols 0..31
      c = wmma_bf16(hA1, ldsB(lds_w1, 96, nt, 1, lane), c);   // h cols 32..63
      c = wmma_bf16(xA,  ldsB(lds_w1, 96, nt, 2, lane), c);   // x cols 0..31
      f1[nt] = c;
    }
    asm volatile("" ::: "memory");
#pragma unroll
    for (int nt = 0; nt < 8; ++nt) stC(stage, 128, nt, f1[nt], lane, true, lds_b1);
    asm volatile("" ::: "memory");

    // GEMM2: f2 = relu(f1) @ W2   (N128, K128 -> 32 wmma)
    v16bf a2[4];
#pragma unroll
    for (int kc = 0; kc < 4; ++kc) a2[kc] = ldsA(stage, 128, kc, lane);
    v8f f2[8];
#pragma unroll
    for (int nt = 0; nt < 8; ++nt) {
      v8f c = zero8();
#pragma unroll
      for (int kc = 0; kc < 4; ++kc)
        c = wmma_bf16(a2[kc], ldsB(lds_w2, 128, nt, kc, lane), c);
      f2[nt] = c;
    }
    asm volatile("" ::: "memory");
#pragma unroll
    for (int nt = 0; nt < 8; ++nt) stC(stage, 128, nt, f2[nt], lane, true, lds_b2);
    asm volatile("" ::: "memory");

    // GEMM3: f3 = relu(f2) @ W3   (N64, K128 -> 16 wmma); bias added at update
    v16bf a3[4];
#pragma unroll
    for (int kc = 0; kc < 4; ++kc) a3[kc] = ldsA(stage, 128, kc, lane);
    v8f f3[4];
#pragma unroll
    for (int nt = 0; nt < 4; ++nt) {
      v8f c = zero8();
#pragma unroll
      for (int kc = 0; kc < 4; ++kc)
        c = wmma_bf16(a3[kc], ldsB(lds_w3, 128, nt, kc, lane), c);
      f3[nt] = c;
    }

    // Diffusion: g1v = relu(h @ D1 + d1)   (N64, K64 -> 8 wmma)
    v8f g1v[4];
#pragma unroll
    for (int nt = 0; nt < 4; ++nt) {
      v8f c = zero8();
      c = wmma_bf16(hA0, ldsB(lds_d1w, 64, nt, 0, lane), c);
      c = wmma_bf16(hA1, ldsB(lds_d1w, 64, nt, 1, lane), c);
      g1v[nt] = c;
    }
    asm volatile("" ::: "memory");
#pragma unroll
    for (int nt = 0; nt < 4; ++nt) stC(stage, 64, nt, g1v[nt], lane, true, lds_d1b);
    asm volatile("" ::: "memory");
    v16bf ga0 = ldsA(stage, 64, 0, lane);
    v16bf ga1 = ldsA(stage, 64, 1, lane);

    // Noise tile is needed now: wait for the TDM copy, then read from LDS.
    __builtin_amdgcn_s_wait_tensorcnt(0);
    asm volatile("" ::: "memory");

    // g2 = g1v @ D2 + d2; g = scale*sigmoid(g2); h += DT*f + sqrt(DT)*g*nz
#pragma unroll
    for (int nt = 0; nt < 4; ++nt) {
      v8f c = zero8();
      c = wmma_bf16(ga0, ldsB(lds_d2w, 64, nt, 0, lane), c);
      c = wmma_bf16(ga1, ldsB(lds_d2w, 64, nt, 1, lane), c);
      float bi  = lds_d2b[nt * 16 + n];
      float sc  = lds_sc[nt * 16 + n];
      float b3v = lds_b3[nt * 16 + n];
#pragma unroll
      for (int v = 0; v < 8; ++v) {
        float gv = c[v] + bi;
        gv = sc / (1.0f + __expf(-gv));                       // scale * sigmoid
        float nz = nzlds[(v + 8 * hk) * H_DIM + nt * 16 + n]; // ds_load from TDM tile
        hc[nt][v] = hc[nt][v] + DT_C * (f3[nt][v] + b3v) + SQRT_DT * gv * nz;
      }
    }
  }

  // ---- readout: out = relu(h @ R1 + r1) @ R2 + r2 ----
#pragma unroll
  for (int nt = 0; nt < 4; ++nt) stC(stage, 64, nt, hc[nt], lane, false, nullptr);
  asm volatile("" ::: "memory");
  v16bf ha0 = ldsA(stage, 64, 0, lane);
  v16bf ha1 = ldsA(stage, 64, 1, lane);
  v8f u0 = zero8(), u1 = zero8();
  u0 = wmma_bf16(ha0, ldsB(lds_r1w, 64, 0, 0, lane), u0);
  u0 = wmma_bf16(ha1, ldsB(lds_r1w, 64, 0, 1, lane), u0);
  u1 = wmma_bf16(ha0, ldsB(lds_r1w, 64, 1, 0, lane), u1);
  u1 = wmma_bf16(ha1, ldsB(lds_r1w, 64, 1, 1, lane), u1);
  asm volatile("" ::: "memory");
  stC(stage, 32, 0, u0, lane, true, lds_r1b);
  stC(stage, 32, 1, u1, lane, true, lds_r1b);
  asm volatile("" ::: "memory");
  v16bf ua = ldsA(stage, 32, 0, lane);
  v8f o = zero8();
  o = wmma_bf16(ua, ldsB(lds_r2w, 32, 0, 0, lane), o);   // N padded to 16; cols 2..15 zero

  if (n < 2) {
    float rb = lds_r2b[n];
#pragma unroll
    for (int v = 0; v < 8; ++v)
      out[(size_t)n * B_TOTAL + r0 + v + 8 * hk] = o[v] + rb;
  }
}

extern "C" void kernel_launch(void* const* d_in, const int* in_sizes, int n_in,
                              void* d_out, int out_size, void* d_ws, size_t ws_size,
                              hipStream_t stream) {
  const float* x_path = (const float*)d_in[0];
  // d_in[1] = t_span: idx == step exactly, unused
  const float* noise  = (const float*)d_in[2];
  const float* W1 = (const float*)d_in[3];  const float* b1 = (const float*)d_in[4];
  const float* W2 = (const float*)d_in[5];  const float* b2 = (const float*)d_in[6];
  const float* W3 = (const float*)d_in[7];  const float* b3 = (const float*)d_in[8];
  const float* D1 = (const float*)d_in[9];  const float* d1 = (const float*)d_in[10];
  const float* D2 = (const float*)d_in[11]; const float* d2 = (const float*)d_in[12];
  const float* scale = (const float*)d_in[13];
  const float* R1 = (const float*)d_in[14]; const float* r1 = (const float*)d_in[15];
  const float* R2 = (const float*)d_in[16]; const float* r2 = (const float*)d_in[17];

  dim3 grid(B_TOTAL / ROWS_PER_WG);   // 64 workgroups
  dim3 block(BLOCK_THREADS);          // 128 threads = 4 wave32
  hipLaunchKernelGGL(nsde_fused_kernel, grid, block, (size_t)SMEM_BYTES, stream,
                     x_path, noise, W1, b1, W2, b2, W3, b3,
                     D1, d1, D2, d2, scale, R1, r1, R2, r2, (float*)d_out);
}
